// SelfAttentionLayer_23579370455265
// MI455X (gfx1250) — matrix operand order repaired and checked by reading, not compiled, not run on hardware
//
#include <hip/hip_runtime.h>

// ---- problem constants -----------------------------------------------------
constexpr int Bn = 64, Sn = 512, En = 512, Hn = 8, Dn = 64;
constexpr int QKVN = 192;                         // Q|K|V columns per head
constexpr float kScale = 0.044194173824159216f;   // 1/sqrt(512)

#if __has_builtin(__builtin_amdgcn_global_load_async_to_lds_b128) && \
    __has_builtin(__builtin_amdgcn_s_wait_asynccnt)
#define USE_ASYNC_LDS 1
#else
#define USE_ASYNC_LDS 0
#endif

// ---- WMMA types ------------------------------------------------------------
typedef __attribute__((ext_vector_type(16))) __bf16        v16bf;
typedef __attribute__((ext_vector_type(8)))  float         v8f;
typedef __attribute__((ext_vector_type(4)))  unsigned int  v4u;

// exact parameter type for the async-LDS builtins (from clang diagnostic):
// '__attribute__((__vector_size__(4 * sizeof(int)))) int' pointers
typedef int v4i_gcc __attribute__((__vector_size__(16)));
typedef __attribute__((address_space(1))) v4i_gcc* as1_v4i_ptr;
typedef __attribute__((address_space(3))) v4i_gcc* as3_v4i_ptr;

union BF16x16 { v16bf v; v4u q[2]; unsigned int u[8]; unsigned short h[16]; };
union BFPack8 { v4u q; unsigned short h[8]; };

__device__ __forceinline__ v8f wmma_bf16(v16bf a, v16bf b, v8f c) {
  return __builtin_amdgcn_wmma_f32_16x16x32_bf16(
      false, a, false, b, (short)0, c, false, false);
}

__device__ __forceinline__ unsigned short f2bf(float f) {
  unsigned int u = __float_as_uint(f);
  u = (u + 0x7FFFu + ((u >> 16) & 1u)) >> 16;
  return (unsigned short)u;
}

__device__ __forceinline__ int a_kpair(int v, int hf) {
  return ((v & 3) << 1) + ((v >> 2) << 4) + (hf << 3);
}

// Fragment-packed tile: 32 lanes x 8 consecutive u32 (256 u32 per 16x32 tile).
__device__ __forceinline__ v16bf load_frag_packed(const unsigned int* tile) {
  BF16x16 f;
  const v4u* p = (const v4u*)(tile + (threadIdx.x & 31) * 8);
  f.q[0] = p[0];
  f.q[1] = p[1];
  return f.v;
}

// Strided A fragment from row-major bf16 source (used only for Q in attn).
__device__ __forceinline__ v16bf load_a_strided(const unsigned short* p, int lda) {
  BF16x16 f;
  const int lr = threadIdx.x & 15;
  const int hf = (threadIdx.x >> 4) & 1;
  const unsigned short* rp = p + (size_t)lr * lda;
#pragma unroll
  for (int v = 0; v < 8; ++v)
    f.u[v] = *(const unsigned int*)(rp + 2 * v + 16 * hf);
  return f.v;
}

// ---- kernel 1: h = x + BE -> bf16 in A-fragment-packed layout ---------------
// hfrag[b][mt(32)][kt(16)][lane(32)][v(8)]  (u32 granules)
__global__ __launch_bounds__(256) void prep_h_kernel(
    const float* __restrict__ x, const float* __restrict__ be,
    unsigned int* __restrict__ hfrag) {
  unsigned int i = blockIdx.x * blockDim.x + threadIdx.x;
  if (i >= (unsigned)Bn * 32 * 16 * 32 * 8) return;
  int v    = i & 7;
  int lane = (i >> 3) & 31;
  int kt   = (i >> 8) & 15;
  int mt   = (i >> 12) & 31;
  int b    = i >> 17;
  int lr = lane & 15, hf = lane >> 4;
  int s  = mt * 16 + lr;
  int e0 = kt * 32 + a_kpair(v, hf);
  size_t xi = ((size_t)b * Sn + s) * En + e0;
  size_t bi = (size_t)s * En + e0;
  unsigned int lo = f2bf(x[xi] + be[bi]);
  unsigned int hi = f2bf(x[xi + 1] + be[bi + 1]);
  hfrag[i] = lo | (hi << 16);
}

// ---- kernel 2: weights -> bf16 B-fragment-packed layouts --------------------
__global__ __launch_bounds__(256) void prep_w_kernel(
    const float* __restrict__ wq, const float* __restrict__ wk,
    const float* __restrict__ wv, const float* __restrict__ wo,
    unsigned int* __restrict__ wpack, unsigned int* __restrict__ wopack) {
  unsigned int i = blockIdx.x * blockDim.x + threadIdx.x;
  if (i < (unsigned)Hn * 16 * 12 * 32 * 8) {
    int v    = i & 7;
    unsigned int t = i >> 3;
    int lane = t & 31;  t >>= 5;
    int nt   = t % 12;  t /= 12;
    int kt   = t % 16;
    int h    = t / 16;
    int lr = lane & 15, hf = lane >> 4;
    int n  = nt * 16 + lr;
    int k0 = kt * 32 + a_kpair(v, hf);
    float f0, f1;
    if (n < 64) {
      f0 = wq[((size_t)h * En + k0) * Dn + n];
      f1 = wq[((size_t)h * En + k0 + 1) * Dn + n];
    } else if (n < 128) {
      f0 = wk[((size_t)h * En + k0) * Dn + (n - 64)];
      f1 = wk[((size_t)h * En + k0 + 1) * Dn + (n - 64)];
    } else {
      f0 = wv[((size_t)h * En + k0) * Dn + (n - 128)];
      f1 = wv[((size_t)h * En + k0 + 1) * Dn + (n - 128)];
    }
    wpack[i] = (unsigned)f2bf(f0) | ((unsigned)f2bf(f1) << 16);
  }
  if (i < (unsigned)16 * 32 * 32 * 8) {
    int v    = i & 7;
    unsigned int t = i >> 3;
    int lane = t & 31;  t >>= 5;
    int nt   = t % 32;
    int kt   = t / 32;
    int lr = lane & 15, hf = lane >> 4;
    int n  = nt * 16 + lr;
    int k0 = kt * 32 + a_kpair(v, hf);
    unsigned int lo = f2bf(wo[(size_t)k0 * En + n]);
    unsigned int hi = f2bf(wo[(size_t)(k0 + 1) * En + n]);
    wopack[i] = lo | (hi << 16);
  }
}

// ---- kernel 3: QKV projection, 2x2 register-blocked -------------------------
// outputs: qk[b][h][s][128] row-major (Q cols 0..63, K cols 64..127)
//          vT[b][h][d(64)][S]  (V transposed; rows contiguous in s)
__global__ __launch_bounds__(256) void qkv_kernel(
    const unsigned int* __restrict__ hfrag,
    const unsigned int* __restrict__ wpack,
    unsigned short* __restrict__ qk,
    unsigned short* __restrict__ vT) {
  int gw = (int)((blockIdx.x * blockDim.x + threadIdx.x) >> 5);
  int ntp = gw % 6;   gw /= 6;    // nt pair: covers nt = 2*ntp, 2*ntp+1
  int mtp = gw % 16;  gw /= 16;   // mt pair: covers mt = 2*mtp, 2*mtp+1
  int h   = gw % Hn;
  int b   = gw / Hn;

  const unsigned int* At0 = hfrag + ((size_t)(b * 32 + 2 * mtp) * 16) * 256;
  const unsigned int* At1 = At0 + 16 * 256;
  const unsigned int* Bt  = wpack + ((size_t)(h * 16) * 12 + 2 * ntp) * 256;

  v8f c00 = {0,0,0,0,0,0,0,0}, c01 = {0,0,0,0,0,0,0,0};
  v8f c10 = {0,0,0,0,0,0,0,0}, c11 = {0,0,0,0,0,0,0,0};
  for (int kt = 0; kt < 16; ++kt) {
    if (kt + 1 < 16) {
      __builtin_prefetch(At0 + (kt + 1) * 256, 0, 1);
      __builtin_prefetch(At1 + (kt + 1) * 256, 0, 1);
      __builtin_prefetch(Bt + (size_t)(kt + 1) * 12 * 256, 0, 1);
    }
    v16bf a0 = load_frag_packed(At0 + kt * 256);
    v16bf a1 = load_frag_packed(At1 + kt * 256);
    v16bf b0 = load_frag_packed(Bt + (size_t)kt * 12 * 256);
    v16bf b1 = load_frag_packed(Bt + (size_t)kt * 12 * 256 + 256);
    c00 = wmma_bf16(a0, b0, c00);
    c01 = wmma_bf16(a0, b1, c01);
    c10 = wmma_bf16(a1, b0, c10);
    c11 = wmma_bf16(a1, b1, c11);
  }

  const int lr = threadIdx.x & 15;
  const int hf = (threadIdx.x >> 4) & 1;
  v8f cs[4] = {c00, c01, c10, c11};
#pragma unroll
  for (int t = 0; t < 4; ++t) {
    int mt = 2 * mtp + (t >> 1);
    int nt = 2 * ntp + (t & 1);
    v8f c = cs[t];
    if (nt < 8) {
      // Q|K -> row-major qk
      unsigned short* Cp =
          qk + (((size_t)(b * Hn + h) * Sn + mt * 16) * 128 + nt * 16);
#pragma unroll
      for (int g = 0; g < 8; ++g)
        Cp[(size_t)(g + hf * 8) * 128 + lr] = f2bf(c[g]);
    } else {
      // V -> transposed vT[d][s]; one b128 store per lane
      BFPack8 pk;
#pragma unroll
      for (int g = 0; g < 8; ++g) pk.h[g] = f2bf(c[g]);
      unsigned short* row =
          vT + ((size_t)(b * Hn + h) * 64 + (nt - 8) * 16 + lr) * Sn;
      *(v4u*)(row + mt * 16 + hf * 8) = pk.q;
    }
  }
}

// ---- kernel 4: fused flash attention, async double-buffered K/V staging -----
__global__ __launch_bounds__(256) void attn_kernel(
    const unsigned short* __restrict__ qk,
    const unsigned short* __restrict__ vT,
    const float* __restrict__ mask,
    unsigned int* __restrict__ aoutp) {
  __shared__ __attribute__((aligned(16))) unsigned int   KtBuf[2][32 * 32]; // [key][d]
  __shared__ __attribute__((aligned(16))) unsigned int   VtBuf[2][64 * 16]; // [d][key]
  __shared__ __attribute__((aligned(16))) unsigned short Pw[8][16 * 32];
  __shared__ __attribute__((aligned(16))) unsigned short Ow[8][16 * 64];

  const int wave = threadIdx.x >> 5;
  const int lr   = threadIdx.x & 15;
  const int hf   = (threadIdx.x >> 4) & 1;
  const int b  = blockIdx.x;
  const int h  = blockIdx.y;
  const int q0 = blockIdx.z * 128 + wave * 16;

  const unsigned short* qkbase = qk + ((size_t)(b * Hn + h) * Sn) * 128;
  const unsigned short* vtbase = vT + ((size_t)(b * Hn + h) * 64) * Sn;

  // per-thread staging chunk coordinates (16B granules)
  const int tid = threadIdx.x;
  const int kr = tid >> 3, kc = tid & 7;   // K: 32 rows x 8 chunks (64 shorts)
  const int vr = tid >> 2, vc = tid & 3;   // V: 64 rows x 4 chunks (32 keys)

  auto stage = [&](int k0, int buf) {
    const unsigned short* ksrc = qkbase + (size_t)(k0 + kr) * 128 + 64 + kc * 8;
    unsigned int* kdst = &KtBuf[buf][kr * 32 + kc * 4];
    const unsigned short* vsrc = vtbase + (size_t)vr * Sn + k0 + vc * 8;
    unsigned int* vdst = &VtBuf[buf][vr * 16 + vc * 4];
#if USE_ASYNC_LDS
    __builtin_amdgcn_global_load_async_to_lds_b128(
        (as1_v4i_ptr)(void*)ksrc, (as3_v4i_ptr)(void*)kdst, 0, 0);
    __builtin_amdgcn_global_load_async_to_lds_b128(
        (as1_v4i_ptr)(void*)vsrc, (as3_v4i_ptr)(void*)vdst, 0, 0);
#else
    *(v4u*)kdst = *(const v4u*)ksrc;
    *(v4u*)vdst = *(const v4u*)vsrc;
#endif
  };

  v16bf qf0 = load_a_strided(qkbase + (size_t)q0 * 128 + 0,  128);
  v16bf qf1 = load_a_strided(qkbase + (size_t)q0 * 128 + 32, 128);

  float m[8], l[8];
  v8f oc[4];
#pragma unroll
  for (int g = 0; g < 8; ++g) { m[g] = -3.0e38f; l[g] = 0.0f; }
#pragma unroll
  for (int cn = 0; cn < 4; ++cn) oc[cn] = (v8f){0, 0, 0, 0, 0, 0, 0, 0};

  stage(0, 0);
  for (int t = 0; t < 16; ++t) {
    const int buf = t & 1;
    const int k0 = t * 32;
    if (t + 1 < 16) {
      stage((t + 1) * 32, buf ^ 1);
#if USE_ASYNC_LDS
      __builtin_amdgcn_s_wait_asynccnt(2);   // this tile's 2 ops retired
#endif
    } else {
#if USE_ASYNC_LDS
      __builtin_amdgcn_s_wait_asynccnt(0);
#endif
    }
    __syncthreads();

    const unsigned int* Kt  = &KtBuf[buf][0];
    const unsigned int* Vt  = &VtBuf[buf][0];

    v8f s0 = {0,0,0,0,0,0,0,0};
    v8f s1 = {0,0,0,0,0,0,0,0};
    {
      BF16x16 kb;
      const v4u* p;
      p = (const v4u*)&Kt[lr * 32 + hf * 8];
      kb.q[0] = p[0]; kb.q[1] = p[1];
      s0 = wmma_bf16(qf0, kb.v, s0);
      p = (const v4u*)&Kt[lr * 32 + 16 + hf * 8];
      kb.q[0] = p[0]; kb.q[1] = p[1];
      s0 = wmma_bf16(qf1, kb.v, s0);
      p = (const v4u*)&Kt[(16 + lr) * 32 + hf * 8];
      kb.q[0] = p[0]; kb.q[1] = p[1];
      s1 = wmma_bf16(qf0, kb.v, s1);
      p = (const v4u*)&Kt[(16 + lr) * 32 + 16 + hf * 8];
      kb.q[0] = p[0]; kb.q[1] = p[1];
      s1 = wmma_bf16(qf1, kb.v, s1);
    }

    float mk0 = mask[(size_t)b * Sn + k0 + lr]      * 1.0e10f;
    float mk1 = mask[(size_t)b * Sn + k0 + 16 + lr] * 1.0e10f;

#pragma unroll
    for (int g = 0; g < 8; ++g) {
      float a0 = s0[g] * kScale - mk0;
      float a1 = s1[g] * kScale - mk1;
      float mx = fmaxf(a0, a1);
#pragma unroll
      for (int off = 1; off < 16; off <<= 1)
        mx = fmaxf(mx, __shfl_xor(mx, off, 32));
      float mnew = fmaxf(m[g], mx);
      float scl = __expf(m[g] - mnew);
      m[g] = mnew;
      float p0 = __expf(a0 - mnew);
      float p1 = __expf(a1 - mnew);
      float rs = p0 + p1;
#pragma unroll
      for (int off = 1; off < 16; off <<= 1)
        rs += __shfl_xor(rs, off, 32);
      l[g] = l[g] * scl + rs;
#pragma unroll
      for (int cn = 0; cn < 4; ++cn) oc[cn][g] *= scl;
      Pw[wave][(g + hf * 8) * 32 + lr]      = f2bf(p0);
      Pw[wave][(g + hf * 8) * 32 + 16 + lr] = f2bf(p1);
    }

    BF16x16 pf;
    {
      const unsigned int* pw32 = (const unsigned int*)&Pw[wave][0];
      pf.q[0] = *(const v4u*)&pw32[lr * 16 + 8 * hf];
      pf.q[1] = *(const v4u*)&pw32[lr * 16 + 8 * hf + 4];
    }

#pragma unroll
    for (int cn = 0; cn < 4; ++cn) {
      BF16x16 vb;
      const v4u* p = (const v4u*)&Vt[(cn * 16 + lr) * 16 + hf * 8];
      vb.q[0] = p[0]; vb.q[1] = p[1];
      oc[cn] = wmma_bf16(pf.v, vb.v, oc[cn]);
    }
    __syncthreads();
  }

  // normalize; bounce C-layout -> A-layout through LDS; store packed fragments
#pragma unroll
  for (int g = 0; g < 8; ++g) {
    float inv = 1.0f / l[g];
#pragma unroll
    for (int cn = 0; cn < 4; ++cn)
      Ow[wave][(g + hf * 8) * 64 + cn * 16 + lr] = f2bf(oc[cn][g] * inv);
  }
  const unsigned int* ow32 = (const unsigned int*)&Ow[wave][0];
  int mtg = (b * Sn + q0) >> 4;
  int lane = threadIdx.x & 31;
#pragma unroll
  for (int ktl = 0; ktl < 2; ++ktl) {
    v4u r0 = *(const v4u*)&ow32[lr * 32 + ktl * 16 + 4 * hf];
    v4u r1 = *(const v4u*)&ow32[lr * 32 + ktl * 16 + 8 + 4 * hf];
    v4u* dst = (v4u*)(aoutp + ((size_t)mtg * 16 + (h * 2 + ktl)) * 256 + lane * 8);
    dst[0] = r0;
    dst[1] = r1;
  }
}

// ---- kernel 5: output projection, 2x2 register-blocked ----------------------
__global__ __launch_bounds__(256) void out_kernel(
    const unsigned int* __restrict__ aoutp,
    const unsigned int* __restrict__ wopack,
    float* __restrict__ out) {
  int gw = (int)((blockIdx.x * blockDim.x + threadIdx.x) >> 5);
  int ntp = gw % 16;         // nt = 2*ntp, 2*ntp+1 (of 32)
  int mtp = gw / 16;         // mt = 2*mtp, 2*mtp+1 (of 2048)

  const unsigned int* At0 = aoutp + (size_t)(2 * mtp) * 16 * 256;
  const unsigned int* At1 = At0 + 16 * 256;

  v8f c00 = {0,0,0,0,0,0,0,0}, c01 = {0,0,0,0,0,0,0,0};
  v8f c10 = {0,0,0,0,0,0,0,0}, c11 = {0,0,0,0,0,0,0,0};
  for (int kt = 0; kt < 16; ++kt) {
    if (kt + 1 < 16) {
      __builtin_prefetch(At0 + (kt + 1) * 256, 0, 1);
      __builtin_prefetch(At1 + (kt + 1) * 256, 0, 1);
      __builtin_prefetch(wopack + (size_t)((kt + 1) * 32 + 2 * ntp) * 256, 0, 1);
    }
    v16bf a0 = load_frag_packed(At0 + kt * 256);
    v16bf a1 = load_frag_packed(At1 + kt * 256);
    v16bf b0 = load_frag_packed(wopack + (size_t)(kt * 32 + 2 * ntp) * 256);
    v16bf b1 = load_frag_packed(wopack + (size_t)(kt * 32 + 2 * ntp + 1) * 256);
    c00 = wmma_bf16(a0, b0, c00);
    c01 = wmma_bf16(a0, b1, c01);
    c10 = wmma_bf16(a1, b0, c10);
    c11 = wmma_bf16(a1, b1, c11);
  }

  const int lr = threadIdx.x & 15;
  const int hf = (threadIdx.x >> 4) & 1;
  v8f cs[4] = {c00, c01, c10, c11};
#pragma unroll
  for (int t = 0; t < 4; ++t) {
    int mt = 2 * mtp + (t >> 1);
    int nt = 2 * ntp + (t & 1);
    v8f c = cs[t];
    float* Cp = out + (size_t)mt * 16 * En + nt * 16;
#pragma unroll
    for (int g = 0; g < 8; ++g)
      Cp[(size_t)(g + hf * 8) * En + lr] = c[g];
  }
}

// ---- launch -----------------------------------------------------------------
extern "C" void kernel_launch(void* const* d_in, const int* in_sizes, int n_in,
                              void* d_out, int out_size, void* d_ws, size_t ws_size,
                              hipStream_t stream) {
  (void)in_sizes; (void)n_in; (void)out_size; (void)ws_size;
  const float* x    = (const float*)d_in[0];
  const float* mask = (const float*)d_in[1];
  const float* WQ   = (const float*)d_in[2];
  const float* WK   = (const float*)d_in[3];
  const float* WV   = (const float*)d_in[4];
  const float* BE   = (const float*)d_in[5];
  const float* WO   = (const float*)d_in[6];
  float* out = (float*)d_out;

  char* ws = (char*)d_ws;
  unsigned int*   hfrag  = (unsigned int*)(ws);                    // 33,554,432 B
  unsigned int*   wpack  = (unsigned int*)(ws + 33554432);         //  1,572,864 B
  unsigned int*   wopack = (unsigned int*)(ws + 35127296);         //    524,288 B
  unsigned short* qk     = (unsigned short*)(ws + 35651584);       // 67,108,864 B
  unsigned short* vT     = (unsigned short*)(ws + 102760448);      // 33,554,432 B
  unsigned int*   aoutp  = (unsigned int*)(ws + 136314880);        // 33,554,432 B

  prep_h_kernel<<<32768, 256, 0, stream>>>(x, BE, hfrag);
  prep_w_kernel<<<1536, 256, 0, stream>>>(WQ, WK, WV, WO, wpack, wopack);

  // B*H * 16 mt-pairs * 6 nt-pairs waves, 8 waves/block
  qkv_kernel<<<(Bn * Hn * 16 * 6) / 8, 256, 0, stream>>>(hfrag, wpack, qk, vT);

  dim3 ag(Bn, Hn, Sn / 128);
  attn_kernel<<<ag, 256, 0, stream>>>(qk, vT, mask, aoutp);

  // 1024 mt-pairs * 16 nt-pairs waves, 8 waves/block
  out_kernel<<<(1024 * 16) / 8, 256, 0, stream>>>(aoutp, wopack, out);
}
